// CoreFinder_29643864277126
// MI455X (gfx1250) — matrix operand order repaired and checked by reading, not compiled, not run on hardware
//
#include <hip/hip_runtime.h>
#include <cstdint>

#define TPB 256

namespace {

constexpr int kNV = 20000;       // variables
constexpr int kNC = 84000;       // clauses
constexpr int kNL = 2 * kNV;     // literals
constexpr int kNE = kNC * 3;     // edges (LPC = 3)
constexpr int kNG = 64;          // graphs
constexpr int kRounds = 4;
constexpr int kNVp = 20032;      // kNV rounded up to 64 (GEMM row pad)
constexpr int kNCp = 84032;      // kNC rounded up to 64

typedef __attribute__((ext_vector_type(16))) _Float16 v16h;
typedef __attribute__((ext_vector_type(8)))  float    v8f;
typedef __attribute__((ext_vector_type(4)))  float    v4f;

__host__ __device__ inline int cdivi(long a, long b) { return (int)((a + b - 1) / b); }

__device__ __forceinline__ float softplusf(float x) {
  return x > 0.f ? x + log1pf(expf(-x)) : log1pf(expf(x));
}
__device__ __forceinline__ float sigmf(float x) { return 1.f / (1.f + expf(-x)); }

__device__ __forceinline__ unsigned hu(unsigned x) {
  x ^= x >> 16; x *= 0x7feb352du; x ^= x >> 15; x *= 0x846ca68bu; x ^= x >> 16; return x;
}
__device__ __forceinline__ float u01(unsigned s) {
  return ((float)(hu(s) >> 8) + 0.5f) * (1.0f / 16777216.0f);
}
__device__ __forceinline__ float gaussf(unsigned s) {
  float a = u01(s);
  float b = u01(s ^ 0x9e3779b9u);
  return sqrtf(fmaxf(-2.f * logf(a), 0.f)) * __cosf(6.28318530718f * b);
}
__device__ __forceinline__ unsigned mix4(unsigned seed, unsigned tag, unsigned r, unsigned c) {
  return hu(seed * 0x9e3779b9u ^ tag * 0x85ebca6bu ^ r * 0xc2b2ae35u ^ (c + 0x27d4eb2fu));
}

// ---------------------------------------------------------------------------
// Weight pre-pack: W[K,N] f32 -> fragment-layout f16, zero-padded to K%32==0.
// Unit u = (kt*NT + j)*32 + lane ; each unit = 16 contiguous halves.
// B-fragment mapping: lane = n + 16*hih holds k = kt*32 + 8*hih + h (h<8), kt*32 + 16 + 8*hih + (h-8) (h>=8).
// ---------------------------------------------------------------------------
__global__ void k_packW(const float* __restrict__ W, _Float16* __restrict__ Wp,
                        int K, int N) {
  int u = blockIdx.x * blockDim.x + threadIdx.x;
  int nkt = (K + 31) / 32;
  int NT = N / 16;
  int total = nkt * NT * 32;
  if (u >= total) return;
  int lane = u & 31;
  int j = (u >> 5) % NT;
  int kt = (u >> 5) / NT;
  int n = j * 16 + (lane & 15);
  int hih = (lane >> 4) & 1;
  v16h out;
#pragma unroll
  for (int h = 0; h < 16; ++h) {
    int k = kt * 32 + (h < 8 ? 8 * hih + h : 16 + 8 * hih + (h - 8));
    float v = (k < K) ? W[(long)k * N + n] : 0.f;
    out[h] = (_Float16)v;
  }
  *(v16h*)(Wp + (size_t)u * 16) = out;
}

// ---------------------------------------------------------------------------
// GEMM: C[M,N] = act(A * W + bias), N = NT*16, W pre-packed f16 fragments.
// Requirements (guaranteed by the launcher):
//  - Kp multiple of 32, lda multiple of 4 floats, A 16B-aligned
//  - A buffer has >= ceil(M/64)*64 rows (pad rows readable; results discarded)
//  - K-pad columns of A are zero
// Block = 128 threads (4 waves), tile = 64 rows x N. Double-buffered A stage
// in fragment layout -> 1 barrier / K-tile, all b128 traffic.
// ---------------------------------------------------------------------------
template <int NT>
__global__ void k_gemm(const float* __restrict__ A, int M, int lda, int Kp,
                       const _Float16* __restrict__ Wp, const float* __restrict__ bias,
                       float* __restrict__ Cout, int ldc, int act, const int* gdone) {
  if (gdone && *gdone) return;
  __shared__ __align__(32) _Float16 As[2][4 * 32 * 16];  // [buf][wave][lane][16]
  const int tid  = threadIdx.x;
  const int wave = tid >> 5;
  const int lane = tid & 31;
  const int row0 = blockIdx.x * 64;
  const int m   = lane & 15;
  const int hih = (lane >> 4) & 1;

  // staging role: thread t handles row rr = t>>1, K-half-select hh = t&1
  const int rr = tid >> 1, hh = tid & 1;
  const int gr = row0 + rr;
  const int sOff = (((rr >> 4) * 32) + ((rr & 15) + 16 * hh)) * 16;
  const int fOff = ((wave * 32) + lane) * 16;
  // 32B-aligned: base 256B, lda mult of 32 floats (>=16B), 8*hh = 32B
  const v4f* aRow4 = (const v4f*)(A + (long)gr * lda + 8 * hh);

  v8f zero = {0.f, 0.f, 0.f, 0.f, 0.f, 0.f, 0.f, 0.f};
  v8f acc[NT];
#pragma unroll
  for (int j = 0; j < NT; ++j) acc[j] = zero;

  const int nkt = Kp >> 5;
  for (int kt = 0; kt < nkt; ++kt) {
    // A lane-fragment: halves 0..7 = k {kt*32+8hh+0..7},
    //                  halves 8..15 = k {kt*32+16+8hh+0..7}
    v4f a0 = aRow4[kt * 8 + 0];
    v4f a1 = aRow4[kt * 8 + 1];
    v4f a2 = aRow4[kt * 8 + 4];
    v4f a3 = aRow4[kt * 8 + 5];
    v16h pack;
#pragma unroll
    for (int q = 0; q < 4; ++q) {
      pack[q]      = (_Float16)a0[q];
      pack[4 + q]  = (_Float16)a1[q];
      pack[8 + q]  = (_Float16)a2[q];
      pack[12 + q] = (_Float16)a3[q];
    }
    _Float16* buf = As[kt & 1];
    *(v16h*)(buf + sOff) = pack;   // 2x ds_store_b128
    __syncthreads();
    v16h af = *(const v16h*)(buf + fOff);  // 2x ds_load_b128
#pragma unroll
    for (int j = 0; j < NT; ++j) {
      v16h bf = *(const v16h*)(Wp + ((size_t)(kt * NT + j) * 32 + lane) * 16);
      acc[j] = __builtin_amdgcn_wmma_f32_16x16x32_f16(false, af, false, bf,
                                                      (short)0, acc[j], false, false);
    }
  }

  // D layout: lane l, vgpr v -> col = l&15, row = v + 8*((l>>4)&1)
#pragma unroll
  for (int j = 0; j < NT; ++j) {
#pragma unroll
    for (int vv = 0; vv < 8; ++vv) {
      int row = row0 + wave * 16 + vv + 8 * hih;
      int col = j * 16 + m;
      if (row < M) {
        float val = acc[j][vv] + bias[col];
        if (act) val = val > 0.f ? val : 0.2f * val;
        Cout[(long)row * ldc + col] = val;
      }
    }
  }
}

// --------------------------- utility kernels -------------------------------
__global__ void k_zero(float* p, long n) {
  long i = (long)blockIdx.x * blockDim.x + threadIdx.x;
  if (i < n) p[i] = 0.f;
}
__global__ void k_fill(float* p, float v, long n) {
  long i = (long)blockIdx.x * blockDim.x + threadIdx.x;
  if (i < n) p[i] = v;
}
__global__ void k_seti(int* p, int v) { *p = v; }

__global__ void k_degree(const int* __restrict__ el, float* deg) {
  int e = blockIdx.x * blockDim.x + threadIdx.x;
  if (e < kNE) atomicAdd(&deg[el[e]], 1.f);
}
__global__ void k_weights(const float* __restrict__ deg, float* dw, float* vdw) {
  int i = blockIdx.x * blockDim.x + threadIdx.x;
  if (i < kNL) {
    dw[i] = rsqrtf(fmaxf(deg[i], 1.f));
    if (i < kNV) vdw[i] = 4.f * rsqrtf(fmaxf(deg[i] + deg[i + kNV], 1.f));
  }
}
__global__ void k_count(const int* __restrict__ idx, float* cnt, int n) {
  int i = blockIdx.x * blockDim.x + threadIdx.x;
  if (i < n) atomicAdd(&cnt[idx[i]], 1.f);
}
__global__ void k_inv(const float* __restrict__ cnt, float* inv, int n) {
  int i = blockIdx.x * blockDim.x + threadIdx.x;
  if (i < n) inv[i] = cnt[i] > 0.f ? 1.f / cnt[i] : 0.f;
}

// --------------------------- per-round kernels -----------------------------
// v1 buffer has lda 96 (cols 68..95 pre-zeroed, never written here)
__global__ void k_build_v1(const float* __restrict__ V, float* v1, int round,
                           const int* seedp, const int* gd) {
  if (*gd) return;
  long i = (long)blockIdx.x * blockDim.x + threadIdx.x;
  if (i >= (long)kNV * 68) return;
  int r = (int)(i / 68), c = (int)(i % 68);
  v1[(long)r * 96 + c] =
      (c < 64) ? V[(long)r * 64 + c]
               : gaussf(mix4((unsigned)*seedp, 101u + 8u * round, r, c - 64));
}
// v2 buffer has lda 96 (cols 69..95 pre-zeroed)
__global__ void k_build_v2(const float* __restrict__ Cst, const float* __restrict__ mask,
                           float* v2, int round, const int* seedp, const int* gd) {
  if (*gd) return;
  long i = (long)blockIdx.x * blockDim.x + threadIdx.x;
  if (i >= (long)kNC * 69) return;
  int r = (int)(i / 69), c = (int)(i % 69);
  float v;
  if (c < 64)      v = Cst[(long)r * 64 + c];
  else if (c == 64) v = mask[r];
  else              v = gaussf(mix4((unsigned)*seedp, 201u + 8u * round, r, c - 65));
  v2[(long)r * 96 + c] = v;
}

// cl = exp(-sum softplus(+-vq)) * cq ; grad wrt cq = exp(-s)
__global__ void k_clause_val(const float* __restrict__ VQ, const float* __restrict__ CQ,
                             const int* __restrict__ el, float* CL, float* CG,
                             const int* gd) {
  if (*gd) return;
  long i = (long)blockIdx.x * blockDim.x + threadIdx.x;
  if (i >= (long)kNC * 64) return;
  int c = (int)(i >> 6), f = (int)(i & 63);
  float s = 0.f;
#pragma unroll
  for (int j = 0; j < 3; ++j) {
    int lit = el[c * 3 + j];
    float x = (lit < kNV) ? VQ[(long)lit * 64 + f] : -VQ[(long)(lit - kNV) * 64 + f];
    s += softplusf(x);
  }
  float e = expf(-s);
  CG[i] = e;
  CL[i] = e * CQ[i];
}

// variables_grad[v,f] += -cl[c,f] * sigmoid(x_e) * sign_e
__global__ void k_scatter_vgrad(const float* __restrict__ CL, const float* __restrict__ VQ,
                                const int* __restrict__ el, float* VGRAD, const int* gd) {
  if (*gd) return;
  long i = (long)blockIdx.x * blockDim.x + threadIdx.x;
  if (i >= (long)kNE * 64) return;
  int e = (int)(i >> 6), f = (int)(i & 63);
  int c = e / 3;
  int lit = el[e];
  int v = (lit < kNV) ? lit : lit - kNV;
  float sign = (lit < kNV) ? 1.f : -1.f;
  float x = sign * VQ[(long)v * 64 + f];
  atomicAdd(&VGRAD[(long)v * 64 + f], -CL[(long)c * 64 + f] * sigmf(x) * sign);
}

// clause_unit = concat(clause_state, 4*cl, cgrad) * mask  -> [NC,192]
__global__ void k_build_cu(const float* __restrict__ Cst, const float* __restrict__ CL,
                           const float* __restrict__ CG, const float* __restrict__ mask,
                           float* CU, const int* gd) {
  if (*gd) return;
  long i = (long)blockIdx.x * blockDim.x + threadIdx.x;
  if (i >= (long)kNC * 192) return;
  int c = (int)(i / 192), k = (int)(i % 192);
  float m = mask[c];
  float v;
  if (k < 64)       v = Cst[(long)c * 64 + k];
  else if (k < 128) v = 4.f * CL[(long)c * 64 + (k - 64)];
  else              v = CG[(long)c * 64 + (k - 128)];
  CU[i] = v * m;
}

// variables_loss scatter: VL[lit,f] += CD[c, f]  (first 64 cols of CD, ld 128)
__global__ void k_scatter_vl(const float* __restrict__ CD, const int* __restrict__ el,
                             float* VL, const int* gd) {
  if (*gd) return;
  long i = (long)blockIdx.x * blockDim.x + threadIdx.x;
  if (i >= (long)kNE * 64) return;
  int e = (int)(i >> 6), f = (int)(i & 63);
  int c = e / 3;
  atomicAdd(&VL[(long)el[e] * 64 + f], CD[(long)c * 128 + f]);
}

// ------------------------------ pair_norm ----------------------------------
__global__ void k_segsum(const float* __restrict__ X, int ldx, const int* __restrict__ idx,
                         float* sums, int n, int D, const int* gd) {
  if (*gd) return;
  long i = (long)blockIdx.x * blockDim.x + threadIdx.x;
  if (i >= (long)n * D) return;
  int r = (int)(i / D), f = (int)(i % D);
  atomicAdd(&sums[(long)idx[r] * D + f], X[(long)r * ldx + f]);
}
__global__ void k_pn_center(const float* __restrict__ X, int ldx, const int* __restrict__ idx,
                            const float* __restrict__ sums, const float* __restrict__ inv,
                            float* XC, float* varg, int n, int D, const int* gd) {
  if (*gd) return;
  int r = blockIdx.x * blockDim.x + threadIdx.x;
  if (r >= n) return;
  int g = idx[r];
  float iv = inv[g];
  float acc = 0.f;
  for (int f = 0; f < D; ++f) {
    float xc = X[(long)r * ldx + f] - sums[(long)g * D + f] * iv * iv;
    XC[(long)r * D + f] = xc;
    acc += xc * xc;
  }
  atomicAdd(&varg[g], acc / (float)D);
}
__global__ void k_pn_finish(const float* __restrict__ XC, const int* __restrict__ idx,
                            const float* __restrict__ varg, const float* __restrict__ inv,
                            const float* __restrict__ oldst, float* out,
                            float scale, float resid, int n, int D, const int* gd) {
  if (*gd) return;
  long i = (long)blockIdx.x * blockDim.x + threadIdx.x;
  if (i >= (long)n * D) return;
  int r = (int)(i / D);
  int g = idx[r];
  float iv = inv[g];
  float fac = rsqrtf(varg[g] * iv * iv + 1e-6f);
  out[i] = XC[i] * fac * scale + resid * oldst[i];
}

// unit = concat(vgrad*vdw, variables, vpos*dw, vneg*dw) -> [NV,256]
__global__ void k_build_unit(const float* __restrict__ VGRAD, const float* __restrict__ vdw,
                             const float* __restrict__ V, const float* __restrict__ VL,
                             const float* __restrict__ dw, float* UNIT, const int* gd) {
  if (*gd) return;
  long i = (long)blockIdx.x * blockDim.x + threadIdx.x;
  if (i >= (long)kNV * 256) return;
  int r = (int)(i >> 8), k = (int)(i & 255);
  float v;
  if (k < 64)       v = VGRAD[(long)r * 64 + k] * vdw[r];
  else if (k < 128) v = V[(long)r * 64 + (k - 64)];
  else if (k < 192) v = VL[(long)r * 64 + (k - 128)] * dw[r];
  else              v = VL[(long)(r + kNV) * 64 + (k - 192)] * dw[r + kNV];
  UNIT[i] = v;
}

// final 64->1 layer + logistic noise
__global__ void k_logits(const float* __restrict__ VOH, const float* __restrict__ W,
                         const float* __restrict__ b, float* logits, int round,
                         const int* seedp, const int* gd) {
  if (*gd) return;
  int v = blockIdx.x * blockDim.x + threadIdx.x;
  if (v >= kNV) return;
  float s = b[0];
  for (int k = 0; k < 64; ++k) s += VOH[(long)v * 64 + k] * W[k];
  const float eps = 1e-5f;
  float u = eps + (1.f - 2.f * eps) * u01(mix4((unsigned)*seedp, 301u + 8u * round, v, 0));
  logits[v] = s + logf(u / (1.f - u));
}

__global__ void k_clause_loss(const float* __restrict__ logits, const float* __restrict__ mask,
                              const int* __restrict__ el, const int* __restrict__ cg,
                              float* pg, int* sat, const int* gd) {
  if (*gd) return;
  int c = blockIdx.x * blockDim.x + threadIdx.x;
  if (c >= kNC) return;
  float s = 0.f;
  bool any = false;
#pragma unroll
  for (int j = 0; j < 3; ++j) {
    int lit = el[c * 3 + j];
    float x = (lit < kNV) ? logits[lit] : -logits[lit - kNV];
    s += softplusf(x);
    any = any || (x > 0.f);
  }
  float cv = expf(-s) * mask[c];
  float pc = cv * -logf(1.f - cv + 1e-10f);
  atomicAdd(&pg[cg[c]], pc);
  if (!any) atomicAnd(sat, 0);
}

__global__ void k_round_finish(const float* __restrict__ pg, float* loss, int* done,
                               int* step, const int* sat, int round) {
  int t = threadIdx.x;
  int d = *done;
  if (!d && t < kNG) loss[t] += sqrtf(pg[t] + 1e-6f) - sqrtf(1e-6f);
  __syncthreads();
  if (t == 0 && !d) {
    *step = round;
    if (*sat) *done = 1;
  }
}

__global__ void k_out(const float* __restrict__ logits, const float* __restrict__ loss,
                      const int* __restrict__ step, float* out) {
  int i = blockIdx.x * blockDim.x + threadIdx.x;
  if (i < kNV)                out[i] = logits[i];
  else if (i < kNV + kNG)     out[i] = loss[i - kNV] * (1.f / (float)kRounds);
  else if (i == kNV + kNG)    out[i] = (float)(*step);
}

}  // namespace

// ---------------------------------------------------------------------------
extern "C" void kernel_launch(void* const* d_in, const int* in_sizes, int n_in,
                              void* d_out, int out_size, void* d_ws, size_t ws_size,
                              hipStream_t stream) {
  (void)in_sizes; (void)n_in; (void)out_size; (void)ws_size;

  const int*   edge_lit = (const int*)d_in[0];
  const int*   var_gidx = (const int*)d_in[2];
  const int*   cls_gidx = (const int*)d_in[3];
  const float* mask     = (const float*)d_in[4];
  // params (dict-insertion-order leaf flattening)
  const float* vq_W0 = (const float*)d_in[5];   const float* vq_b0 = (const float*)d_in[6];
  const float* vq_W1 = (const float*)d_in[7];   const float* vq_b1 = (const float*)d_in[8];
  const float* cq_W0 = (const float*)d_in[9];   const float* cq_b0 = (const float*)d_in[10];
  const float* cq_W1 = (const float*)d_in[11];  const float* cq_b1 = (const float*)d_in[12];
  const float* cm_W0 = (const float*)d_in[13];  const float* cm_b0 = (const float*)d_in[14];
  const float* cm_W1 = (const float*)d_in[15];  const float* cm_b1 = (const float*)d_in[16];
  const float* ug_W0 = (const float*)d_in[17];  const float* ug_b0 = (const float*)d_in[18];
  const float* ug_W1 = (const float*)d_in[19];  const float* ug_b1 = (const float*)d_in[20];
  const float* ug_W2 = (const float*)d_in[21];  const float* ug_b2 = (const float*)d_in[22];
  const float* vo_W0 = (const float*)d_in[23];  const float* vo_b0 = (const float*)d_in[24];
  const float* vo_W1 = (const float*)d_in[25];  const float* vo_b1 = (const float*)d_in[26];
  const int*   seedp = (const int*)d_in[28];

  // ---- workspace bump allocator (f32 units, 64-elem = 256B aligned) ----
  float* base = (float*)d_ws;
  size_t off = 0;
  auto A = [&](size_t n) -> float* {
    float* r = base + off;
    off += (n + 63) & ~(size_t)63;
    return r;
  };
  auto AH = [&](size_t nh) -> _Float16* { return (_Float16*)A((nh + 1) / 2); };

  // GEMM-input buffers get padded rows (kNVp/kNCp); K-pad cols (v1,v2) zeroed once.
  float* V      = A((size_t)kNVp * 64);   // GEMM input (vo L0); rows >= kNV pad
  float* Cst    = A((size_t)kNC * 64);
  float* logits = A(kNV);
  float* loss   = A(kNG);
  float* litdeg = A(kNL);
  float* dw     = A(kNL);
  float* vdw    = A(kNV);
  float* ccnt   = A(kNG); float* vcnt = A(kNG);
  float* inv_c  = A(kNG); float* inv_v = A(kNG);
  float* v1     = A((size_t)kNVp * 96);   // lda 96, cols 68..95 zero
  float* v2     = A((size_t)kNCp * 96);   // lda 96, cols 69..95 zero
  float* H64    = A((size_t)kNCp * 64);   // shared hidden (vq/cq L0); also VOH
  float* VQ     = A((size_t)kNV * 64);
  float* CQ     = A((size_t)kNC * 64);
  float* CL     = A((size_t)kNC * 64);
  float* CG     = A((size_t)kNC * 64);
  float* CU     = A((size_t)kNCp * 192);  // also XCc after clause_mlp
  float* CH     = A((size_t)kNCp * 128);
  float* CD     = A((size_t)kNC * 128);
  float* VGRAD  = A((size_t)kNV * 64);
  float* VL     = A((size_t)kNL * 64);
  float* UNIT   = A((size_t)kNVp * 256);  // also XCv after update_gate
  float* UH1    = A((size_t)kNVp * 128);
  float* UH2    = A((size_t)kNVp * 128);
  float* UGO    = A((size_t)kNV * 64);
  float* meang  = A((size_t)kNG * 64);
  float* varg   = A(kNG);
  float* pg     = A(kNG);
  int*   done   = (int*)A(16);
  int*   step   = done + 1;
  int*   sat    = done + 2;
  float* XCc    = CU;    // reuse
  float* XCv    = UNIT;  // reuse
  float* VOH    = H64;   // reuse

  // packed f16 weights (fragment layout, K padded to 32)
  _Float16* vq0p = AH((size_t)96  * 64);
  _Float16* vq1p = AH((size_t)64  * 64);
  _Float16* cq0p = AH((size_t)96  * 64);
  _Float16* cq1p = AH((size_t)64  * 64);
  _Float16* cm0p = AH((size_t)192 * 128);
  _Float16* cm1p = AH((size_t)128 * 128);
  _Float16* ug0p = AH((size_t)256 * 128);
  _Float16* ug1p = AH((size_t)128 * 128);
  _Float16* ug2p = AH((size_t)128 * 64);
  _Float16* vo0p = AH((size_t)64  * 64);

  auto Z = [&](float* p, long n) { k_zero<<<cdivi(n, TPB), TPB, 0, stream>>>(p, n); };
  auto PACK = [&](const float* W, _Float16* Wp, int K, int N) {
    int units = ((K + 31) / 32) * (N / 16) * 32;
    k_packW<<<cdivi(units, TPB), TPB, 0, stream>>>(W, Wp, K, N);
  };

  // ---- one-time setup ----
  Z(litdeg, kNL); Z(ccnt, kNG); Z(vcnt, kNG); Z(loss, kNG); Z(logits, kNV);
  Z(v1, (long)kNVp * 96);   // establishes zero K-pad columns (68..95)
  Z(v2, (long)kNCp * 96);   // establishes zero K-pad columns (69..95)
  Z(V, (long)kNVp * 64);
  k_fill<<<cdivi((long)kNV * 64, TPB), TPB, 0, stream>>>(V, 1.f, (long)kNV * 64);
  k_fill<<<cdivi((long)kNC * 64, TPB), TPB, 0, stream>>>(Cst, 1.f, (long)kNC * 64);
  k_seti<<<1, 1, 0, stream>>>(done, 0);
  k_seti<<<1, 1, 0, stream>>>(step, 0);
  k_degree<<<cdivi(kNE, TPB), TPB, 0, stream>>>(edge_lit, litdeg);
  k_weights<<<cdivi(kNL, TPB), TPB, 0, stream>>>(litdeg, dw, vdw);
  k_count<<<cdivi(kNC, TPB), TPB, 0, stream>>>(cls_gidx, ccnt, kNC);
  k_count<<<cdivi(kNV, TPB), TPB, 0, stream>>>(var_gidx, vcnt, kNV);
  k_inv<<<1, kNG, 0, stream>>>(ccnt, inv_c, kNG);
  k_inv<<<1, kNG, 0, stream>>>(vcnt, inv_v, kNG);

  PACK(vq_W0, vq0p, 68, 64);   PACK(vq_W1, vq1p, 64, 64);
  PACK(cq_W0, cq0p, 69, 64);   PACK(cq_W1, cq1p, 64, 64);
  PACK(cm_W0, cm0p, 192, 128); PACK(cm_W1, cm1p, 128, 128);
  PACK(ug_W0, ug0p, 256, 128); PACK(ug_W1, ug1p, 128, 128);
  PACK(ug_W2, ug2p, 128, 64);  PACK(vo_W0, vo0p, 64, 64);

  const int gV = cdivi(kNV, 64), gC = cdivi(kNC, 64);

  for (int r = 0; r < kRounds; ++r) {
    Z(VGRAD, (long)kNV * 64); Z(VL, (long)kNL * 64); Z(pg, kNG);
    k_seti<<<1, 1, 0, stream>>>(sat, 1);

    k_build_v1<<<cdivi((long)kNV * 68, TPB), TPB, 0, stream>>>(V, v1, r, seedp, done);
    k_build_v2<<<cdivi((long)kNC * 69, TPB), TPB, 0, stream>>>(Cst, mask, v2, r, seedp, done);

    // variables_query / clauses_query MLPs (WMMA)
    k_gemm<4><<<gV, 128, 0, stream>>>(v1, kNV, 96, 96, vq0p, vq_b0, H64, 64, 1, done);
    k_gemm<4><<<gV, 128, 0, stream>>>(H64, kNV, 64, 64, vq1p, vq_b1, VQ, 64, 0, done);
    k_gemm<4><<<gC, 128, 0, stream>>>(v2, kNC, 96, 96, cq0p, cq_b0, H64, 64, 1, done);
    k_gemm<4><<<gC, 128, 0, stream>>>(H64, kNC, 64, 64, cq1p, cq_b1, CQ, 64, 0, done);

    k_clause_val<<<cdivi((long)kNC * 64, TPB), TPB, 0, stream>>>(VQ, CQ, edge_lit, CL, CG, done);
    k_scatter_vgrad<<<cdivi((long)kNE * 64, TPB), TPB, 0, stream>>>(CL, VQ, edge_lit, VGRAD, done);
    k_build_cu<<<cdivi((long)kNC * 192, TPB), TPB, 0, stream>>>(Cst, CL, CG, mask, CU, done);

    // clause_mlp (WMMA)
    k_gemm<8><<<gC, 128, 0, stream>>>(CU, kNC, 192, 192, cm0p, cm_b0, CH, 128, 1, done);
    k_gemm<8><<<gC, 128, 0, stream>>>(CH, kNC, 128, 128, cm1p, cm_b1, CD, 128, 0, done);

    k_scatter_vl<<<cdivi((long)kNE * 64, TPB), TPB, 0, stream>>>(CD, edge_lit, VL, done);

    // clause pair_norm on CD[:,64:128] -> Cst
    Z(meang, (long)kNG * 64); Z(varg, kNG);
    k_segsum<<<cdivi((long)kNC * 64, TPB), TPB, 0, stream>>>(CD + 64, 128, cls_gidx, meang, kNC, 64, done);
    k_pn_center<<<cdivi(kNC, TPB), TPB, 0, stream>>>(CD + 64, 128, cls_gidx, meang, inv_c, XCc, varg, kNC, 64, done);
    k_pn_finish<<<cdivi((long)kNC * 64, TPB), TPB, 0, stream>>>(XCc, cls_gidx, varg, inv_c, Cst, Cst,
                                                                0.25f, 0.1f, kNC, 64, done);

    k_build_unit<<<cdivi((long)kNV * 256, TPB), TPB, 0, stream>>>(VGRAD, vdw, V, VL, dw, UNIT, done);

    // update_gate (WMMA)
    k_gemm<8><<<gV, 128, 0, stream>>>(UNIT, kNV, 256, 256, ug0p, ug_b0, UH1, 128, 1, done);
    k_gemm<8><<<gV, 128, 0, stream>>>(UH1, kNV, 128, 128, ug1p, ug_b1, UH2, 128, 1, done);
    k_gemm<4><<<gV, 128, 0, stream>>>(UH2, kNV, 128, 128, ug2p, ug_b2, UGO, 64, 0, done);

    // variable pair_norm on UGO -> V
    Z(meang, (long)kNG * 64); Z(varg, kNG);
    k_segsum<<<cdivi((long)kNV * 64, TPB), TPB, 0, stream>>>(UGO, 64, var_gidx, meang, kNV, 64, done);
    k_pn_center<<<cdivi(kNV, TPB), TPB, 0, stream>>>(UGO, 64, var_gidx, meang, inv_v, XCv, varg, kNV, 64, done);
    k_pn_finish<<<cdivi((long)kNV * 64, TPB), TPB, 0, stream>>>(XCv, var_gidx, varg, inv_v, V, V,
                                                                0.25f, 0.1f, kNV, 64, done);

    // variables_output (WMMA L0 + scalar N=1 layer) + logistic noise
    k_gemm<4><<<gV, 128, 0, stream>>>(V, kNV, 64, 64, vo0p, vo_b0, VOH, 64, 1, done);
    k_logits<<<cdivi(kNV, TPB), TPB, 0, stream>>>(VOH, vo_W1, vo_b1, logits, r, seedp, done);

    k_clause_loss<<<cdivi(kNC, TPB), TPB, 0, stream>>>(logits, mask, edge_lit, cls_gidx, pg, sat, done);
    k_round_finish<<<1, 64, 0, stream>>>(pg, loss, done, step, sat, r);
  }

  k_out<<<cdivi(kNV + kNG + 1, TPB), TPB, 0, stream>>>(logits, loss, step, (float*)d_out);
}
